// Deformable_Convolutional_5720896438923
// MI455X (gfx1250) — compile-verified, matrix operand order checked by reading, not run on hardware
//
#include <hip/hip_runtime.h>
#include <math.h>

typedef _Float16 h16;
typedef __attribute__((ext_vector_type(2)))  _Float16 v2h;
typedef __attribute__((ext_vector_type(8)))  _Float16 v8h;
typedef __attribute__((ext_vector_type(16))) _Float16 v16h;
typedef __attribute__((ext_vector_type(8)))  float    v8f;
typedef __attribute__((ext_vector_type(4)))  float    f4;

// Problem constants
#define BSZ   4
#define CIN   128
#define COUT  128
#define HH    96
#define WW    96
#define KK    9
#define KTOT  1152          // CIN*KK
#define KCH   288           // per 32-channel chunk: 32*9
#define SMPS  296           // padded K stride (halves) for im2col panels
#define ROWS  10            // staged rows for deformable sampling (h-4 .. h+5)

// Workspace byte offsets
#define OFF_WT     0u                       // 128*1152 f16  = 294912 B
#define OFF_WOFS   294912u                  // 32*1152 f16   =  73728 B
#define OFF_MPY    368640u                  // 4*9*96*96 f32 = 1327104 B
#define OFF_MPX    1695744u
#define OFF_MMK    3022848u
#define OFF_STATS  4349952u                 // 256 f32

// ---- CDNA5 async global->LDS path (guarded; falls back to sync copies) ----
#if defined(__gfx1250__) && __has_builtin(__builtin_amdgcn_global_load_async_to_lds_b128)
#define ASYNC_LDS 1
#else
#define ASYNC_LDS 0
#endif

typedef __attribute__((__vector_size__(16))) int vi4;
typedef __attribute__((address_space(1))) vi4 gvi4;   // global int4
typedef __attribute__((address_space(3))) vi4 lvi4;   // LDS int4

static __device__ __forceinline__ void wait_async_all() {
#if ASYNC_LDS
#if __has_builtin(__builtin_amdgcn_s_wait_asynccnt)
  __builtin_amdgcn_s_wait_asynccnt(0);
#else
  asm volatile("s_wait_asynccnt 0x0" ::: "memory");
#endif
#endif
}

static __device__ __forceinline__ void copy16_async(const float* g, float* d) {
#if ASYNC_LDS
  __builtin_amdgcn_global_load_async_to_lds_b128((gvi4*)g, (lvi4*)d, 0, 0);
#else
  *(f4*)d = *(const f4*)g;
#endif
}

// Load a 16x32 f16 WMMA A/B fragment (per-lane: K 0..7|8..15 then 16..23|24..31)
// from a 16B-aligned row; caller folds in the (lane<16 ? 0 : 8) selector.
static __device__ __forceinline__ v16h ldfrag(const h16* p) {
  union { v16h v; v8h h[2]; } u;
  u.h[0] = *(const v8h*)(p);
  u.h[1] = *(const v8h*)(p + 16);
  return u.v;
}

static __device__ __forceinline__ v8f wmma_f16(v16h a, v16h b, v8f c) {
  return __builtin_amdgcn_wmma_f32_16x16x32_f16(false, a, false, b, (short)0, c, false, false);
}

// ---------------------------------------------------------------------------
// Kernel 0: transpose + f16-cast weights into [o][K'] with
// K' = (c>>5)*288 + k*32 + (c&31)  (matches the K order of the im2col panels)
// ---------------------------------------------------------------------------
__global__ __launch_bounds__(256) void k_prep(const float* __restrict__ dw,
                                              const float* __restrict__ ow,
                                              h16* __restrict__ wT,
                                              h16* __restrict__ wOfsT) {
  int i = blockIdx.x * 256 + threadIdx.x;
  if (i < COUT * KTOT) {
    int o = i / KTOT, rem = i % KTOT, c = rem / KK, k = rem % KK;
    wT[o * KTOT + (c >> 5) * KCH + k * 32 + (c & 31)] = (h16)dw[i];
  }
  if (i < 32 * KTOT) {
    int o = i / KTOT, rem = i % KTOT, c = rem / KK, k = rem % KK;
    float v = (o < 27) ? ow[i] : 0.f;
    wOfsT[o * KTOT + (c >> 5) * KCH + k * 32 + (c & 31)] = (h16)v;
  }
}

// ---------------------------------------------------------------------------
// Kernel 1: offset conv (3x3, 128->27 padded to 32) via WMMA + meta emit.
// One workgroup per (b,h) row; double-buffered async x staging per 32-ch chunk.
// ---------------------------------------------------------------------------
__global__ __launch_bounds__(256) void k_offs(const float* __restrict__ x,
                                              const h16* __restrict__ wOfsT,
                                              const float* __restrict__ ob,
                                              float* __restrict__ mPy,
                                              float* __restrict__ mPx,
                                              float* __restrict__ mMk) {
  __shared__ __attribute__((aligned(16))) float xt[2][32 * 3 * WW]; //  72 KB
  __shared__ __attribute__((aligned(16))) h16   im[WW * SMPS];      //  55 KB
  __shared__ __attribute__((aligned(16))) float om[32 * WW];        //  12 KB
  const int b = blockIdx.x, h = blockIdx.y;
  const int tid = threadIdx.x, lane = tid & 31, wv = tid >> 5;

  auto stage = [&](int ch, float* dst) {
    for (int v = tid; v < 2304; v += 256) {
      int c = v / 72, rem = v % 72, r = rem / 24, q = rem % 24;
      int y = h - 1 + r;
      float* d = dst + (c * 3 + r) * WW + q * 4;
      if ((unsigned)y < (unsigned)HH) {
        const float* g = x + (((size_t)b * CIN + ch * 32 + c) * HH + y) * WW + q * 4;
        copy16_async(g, d);
      } else {
        f4 zz = {0.f, 0.f, 0.f, 0.f};
        *(f4*)d = zz;
      }
    }
  };

  v8f z = {0.f,0.f,0.f,0.f,0.f,0.f,0.f,0.f};
  v8f acc0 = z, acc1 = z;

  stage(0, xt[0]);
  for (int ch = 0; ch < 4; ++ch) {
    wait_async_all();
    __syncthreads();                        // xt[ch&1] ready, smp consumers done
    if (ch + 1 < 4) stage(ch + 1, xt[(ch + 1) & 1]);
    const float* cur = xt[ch & 1];
    // im2col panel: im[w][tap*32 + c] (f16), zero-padded in x
    for (int pr = tid; pr < KK * WW; pr += 256) {
      int tap = pr / WW, w = pr % WW;
      int ki = tap / 3, kj = tap % 3;
      int xc = w + kj - 1;
      bool vx = (unsigned)xc < (unsigned)WW;
      const float* src = cur + ki * WW + (vx ? xc : 0);
      h16* dst = im + w * SMPS + tap * 32;
      #pragma unroll 4
      for (int c = 0; c < 32; c += 2) {
        float s0 = vx ? src[c * 288] : 0.f;
        float s1 = vx ? src[(c + 1) * 288] : 0.f;
        v2h p = {(h16)s0, (h16)s1};
        *(v2h*)(dst + c) = p;
      }
    }
    __syncthreads();
    if (wv < 6) {
      const int n = lane & 15, sel = (lane < 16) ? 0 : 8;
      const h16* ar  = im + (wv * 16 + n) * SMPS + sel;
      const h16* br0 = wOfsT + (size_t)n * KTOT + ch * KCH + sel;
      const h16* br1 = wOfsT + (size_t)(16 + n) * KTOT + ch * KCH + sel;
      for (int ks = 0; ks < 9; ++ks) {
        v16h af = ldfrag(ar + ks * 32);
        v16h b0 = ldfrag(br0 + ks * 32);
        v16h b1 = ldfrag(br1 + ks * 32);
        acc0 = wmma_f16(af, b0, acc0);
        acc1 = wmma_f16(af, b1, acc1);
      }
    }
  }
  __syncthreads();
  if (wv < 6) {
    const int n = lane & 15, moff = (lane < 16) ? 0 : 8;
    float bo0 = ob[n];                              // n < 16 < 27
    float bo1 = (16 + n < 27) ? ob[16 + n] : 0.f;
    #pragma unroll
    for (int r = 0; r < 8; ++r) {
      om[n * WW + wv * 16 + r + moff]        = acc0[r] + bo0;
      om[(16 + n) * WW + wv * 16 + r + moff] = acc1[r] + bo1;
    }
  }
  __syncthreads();
  // dy=om[2k], dx=om[2k+1], mask=sigmoid(om[18+k]); emit sampling metadata
  for (int pr = tid; pr < KK * WW; pr += 256) {
    int tap = pr / WW, w = pr % WW;
    float dy = om[(2 * tap) * WW + w];
    float dx = om[(2 * tap + 1) * WW + w];
    float mr = om[(18 + tap) * WW + w];
    float py  = dy + (float)(h + tap / 3 - 1);
    float pxv = dx + (float)(w + tap % 3 - 1);
    float mk  = 1.f / (1.f + expf(-mr));
    int mi = ((b * KK + tap) * HH + h) * WW + w;
    mPy[mi] = py; mPx[mi] = pxv; mMk[mi] = mk;
  }
}

// ---------------------------------------------------------------------------
// Kernel 2: main deformable GEMM. One workgroup per (b,h): M=96 (6 tiles),
// N=128 (8 waves x 1 tile), K=1152 in 4 chunks of 288 (32 ch x 9 taps).
// Async double-buffered x-tile staging overlaps the next chunk's 120 KB
// transfer with this chunk's sampling + WMMA sweep.
// ---------------------------------------------------------------------------
__global__ __launch_bounds__(256) void k_main(const float* __restrict__ x,
                                              const h16* __restrict__ wT,
                                              const float* __restrict__ mPy,
                                              const float* __restrict__ mPx,
                                              const float* __restrict__ mMk,
                                              const float* __restrict__ bias,
                                              float* __restrict__ out) {
  __shared__ __attribute__((aligned(16))) float xt[2][32 * ROWS * WW]; // 240 KB
  __shared__ __attribute__((aligned(16))) h16   smp[WW * SMPS];        //  55 KB
  const int b = blockIdx.x, h = blockIdx.y;
  const int tid = threadIdx.x, lane = tid & 31, wv = tid >> 5;

  auto stage = [&](int ch, float* dst) {
    for (int v = tid; v < 7680; v += 256) {
      int c = v / 240, rem = v % 240, r = rem / 24, q = rem % 24;
      int y = h - 4 + r;
      float* d = dst + (c * ROWS + r) * WW + q * 4;
      if ((unsigned)y < (unsigned)HH) {
        const float* g = x + (((size_t)b * CIN + ch * 32 + c) * HH + y) * WW + q * 4;
        copy16_async(g, d);
      } else {
        f4 zz = {0.f, 0.f, 0.f, 0.f};
        *(f4*)d = zz;
      }
    }
  };

  v8f z = {0.f,0.f,0.f,0.f,0.f,0.f,0.f,0.f};
  v8f acc[6];
  #pragma unroll
  for (int i = 0; i < 6; ++i) acc[i] = z;

  stage(0, xt[0]);
  for (int ch = 0; ch < 4; ++ch) {
    wait_async_all();
    __syncthreads();                       // xt[ch&1] ready; smp consumers done
    if (ch + 1 < 4) stage(ch + 1, xt[(ch + 1) & 1]);
    const float* cur = xt[ch & 1];
    // bilinear-sample the im2col panel: smp[w][tap*32+c] (f16, mask folded in)
    for (int pr = tid; pr < KK * WW; pr += 256) {
      int tap = pr / WW, w = pr % WW;
      int mi = ((b * KK + tap) * HH + h) * WW + w;
      float py = mPy[mi], pxv = mPx[mi], mk = mMk[mi];
      float y0f = floorf(py), x0f = floorf(pxv);
      int y0 = (int)y0f, x0 = (int)x0f;
      float wy1 = py - y0f, wx1 = pxv - x0f;
      float wy0 = 1.f - wy1, wx0 = 1.f - wx1;
      int ly0 = y0 - (h - 4), ly1 = ly0 + 1;
      bool vy0 = ((unsigned)y0 < (unsigned)HH) & ((unsigned)ly0 < (unsigned)ROWS);
      bool vy1 = ((unsigned)(y0 + 1) < (unsigned)HH) & ((unsigned)ly1 < (unsigned)ROWS);
      bool vx0 = (unsigned)x0 < (unsigned)WW;
      bool vx1 = (unsigned)(x0 + 1) < (unsigned)WW;
      float w00 = (vy0 && vx0) ? wy0 * wx0 * mk : 0.f;
      float w01 = (vy0 && vx1) ? wy0 * wx1 * mk : 0.f;
      float w10 = (vy1 && vx0) ? wy1 * wx0 * mk : 0.f;
      float w11 = (vy1 && vx1) ? wy1 * wx1 * mk : 0.f;
      int a00 = (vy0 && vx0) ? ly0 * WW + x0     : 0;
      int a01 = (vy0 && vx1) ? ly0 * WW + x0 + 1 : 0;
      int a10 = (vy1 && vx0) ? ly1 * WW + x0     : 0;
      int a11 = (vy1 && vx1) ? ly1 * WW + x0 + 1 : 0;
      h16* dst = smp + w * SMPS + tap * 32;
      #pragma unroll 4
      for (int c = 0; c < 32; c += 2) {
        const float* t0 = cur + c * (ROWS * WW);
        const float* t1 = t0 + (ROWS * WW);
        float s0 = w00 * t0[a00] + w01 * t0[a01] + w10 * t0[a10] + w11 * t0[a11];
        float s1 = w00 * t1[a00] + w01 * t1[a01] + w10 * t1[a10] + w11 * t1[a11];
        v2h p = {(h16)s0, (h16)s1};
        *(v2h*)(dst + c) = p;
      }
    }
    __syncthreads();
    // WMMA sweep: this wave's N-tile (o = 16*wv..) across 6 M-tiles, 9 K-steps
    {
      const int n = lane & 15, sel = (lane < 16) ? 0 : 8;
      const h16* br = wT + (size_t)(wv * 16 + n) * KTOT + ch * KCH + sel;
      const h16* ar = smp + n * SMPS + sel;
      for (int ks = 0; ks < 9; ++ks) {
        v16h bf = ldfrag(br + ks * 32);
        #pragma unroll
        for (int mt = 0; mt < 6; ++mt) {
          v16h af = ldfrag(ar + (mt * 16) * SMPS + ks * 32);
          acc[mt] = wmma_f16(af, bf, acc[mt]);
        }
      }
    }
  }
  __syncthreads();
  // transpose accumulators through LDS (reuse xt[0]) for coalesced stores
  {
    float* tr = xt[0];
    const int n = lane & 15, moff = (lane < 16) ? 0 : 8;
    #pragma unroll
    for (int mt = 0; mt < 6; ++mt)
      #pragma unroll
      for (int r = 0; r < 8; ++r)
        tr[(wv * 16 + n) * WW + mt * 16 + r + moff] = acc[mt][r];
  }
  __syncthreads();
  for (int e = tid; e < COUT * (WW / 4); e += 256) {
    int o = e / 24, q = e % 24;
    f4 v = *(const f4*)(xt[0] + o * WW + q * 4);
    float bz = bias[o];
    f4 r;
    #pragma unroll
    for (int i = 0; i < 4; ++i) r[i] = v[i] + bz;
    *(f4*)(out + (((size_t)b * COUT + o) * HH + h) * WW + q * 4) = r;
  }
}

// ---------------------------------------------------------------------------
// Kernel 3: per-channel mean / rsqrt(var+eps) over (B,H,W)
// ---------------------------------------------------------------------------
__global__ __launch_bounds__(256) void k_stats(const float* __restrict__ o,
                                               float* __restrict__ stats) {
  __shared__ float s1[256], s2[256];
  const int ch = blockIdx.x, tid = threadIdx.x;
  float a = 0.f, sq = 0.f;
  for (int j = tid; j < BSZ * (HH * WW / 4); j += 256) {
    int bb = j / (HH * WW / 4), s = j % (HH * WW / 4);
    f4 v = *(const f4*)(o + ((size_t)(bb * COUT + ch)) * (HH * WW) + s * 4);
    #pragma unroll
    for (int i = 0; i < 4; ++i) { a += v[i]; sq += v[i] * v[i]; }
  }
  s1[tid] = a; s2[tid] = sq;
  __syncthreads();
  for (int st = 128; st > 0; st >>= 1) {
    if (tid < st) { s1[tid] += s1[tid + st]; s2[tid] += s2[tid + st]; }
    __syncthreads();
  }
  if (tid == 0) {
    const float inv = 1.f / (float)(BSZ * HH * WW);
    float mu = s1[0] * inv;
    float var = s2[0] * inv - mu * mu;
    stats[ch] = mu;
    stats[COUT + ch] = rsqrtf(var + 1e-5f);
  }
}

// ---------------------------------------------------------------------------
// Kernel 4: in-place batchnorm + leaky ReLU (0.1)
// ---------------------------------------------------------------------------
__global__ __launch_bounds__(256) void k_bn(float* __restrict__ o,
                                            const float* __restrict__ stats,
                                            const float* __restrict__ gamma,
                                            const float* __restrict__ beta) {
  const int blk = blockIdx.x;
  const int b = blk >> 7, ch = blk & 127;
  const float mu = stats[ch];
  const float g = gamma[ch] * stats[COUT + ch];
  const float bt = beta[ch];
  float* base = o + ((size_t)(b * COUT + ch)) * (HH * WW);
  for (int i = threadIdx.x; i < HH * WW / 4; i += 256) {
    f4 v = *(const f4*)(base + i * 4);
    f4 r;
    #pragma unroll
    for (int q = 0; q < 4; ++q) {
      float t = (v[q] - mu) * g + bt;
      r[q] = t >= 0.f ? t : 0.1f * t;
    }
    *(f4*)(base + i * 4) = r;
  }
}

// ---------------------------------------------------------------------------
extern "C" void kernel_launch(void* const* d_in, const int* in_sizes, int n_in,
                              void* d_out, int out_size, void* d_ws, size_t ws_size,
                              hipStream_t stream) {
  (void)in_sizes; (void)n_in; (void)out_size; (void)ws_size;
  const float* x    = (const float*)d_in[0];
  const float* offw = (const float*)d_in[1];
  const float* offb = (const float*)d_in[2];
  const float* dw   = (const float*)d_in[3];
  const float* db   = (const float*)d_in[4];
  const float* gam  = (const float*)d_in[5];
  const float* bet  = (const float*)d_in[6];
  float* out = (float*)d_out;
  char*  ws  = (char*)d_ws;

  h16*   wT    = (h16*)(ws + OFF_WT);
  h16*   wOfsT = (h16*)(ws + OFF_WOFS);
  float* mPy   = (float*)(ws + OFF_MPY);
  float* mPx   = (float*)(ws + OFF_MPX);
  float* mMk   = (float*)(ws + OFF_MMK);
  float* stats = (float*)(ws + OFF_STATS);

  k_prep <<<576, 256, 0, stream>>>(dw, offw, wT, wOfsT);
  k_offs <<<dim3(BSZ, HH), 256, 0, stream>>>(x, wOfsT, offb, mPy, mPx, mMk);
  k_main <<<dim3(BSZ, HH), 256, 0, stream>>>(x, wT, mPy, mPx, mMk, db, out);
  k_stats<<<COUT, 256, 0, stream>>>(out, stats);
  k_bn   <<<BSZ * COUT, 256, 0, stream>>>(out, stats, gam, bet);
}
